// InteractionBlock_65549790871712
// MI455X (gfx1250) — matrix operand order repaired and checked by reading, not compiled, not run on hardware
//
#include <hip/hip_runtime.h>

#define NNODES 50000
#define NROWSP 50048            // NNODES padded to a multiple of 128 (block row tile)
#define NEDGES 800000
#define HDIM   128
#define ECH    50

typedef __attribute__((ext_vector_type(16))) __bf16 v16bf;
typedef __attribute__((ext_vector_type(8)))  float  v8f;
typedef __attribute__((ext_vector_type(4)))  __bf16 v4bf;

static __device__ __forceinline__ v8f wmma_bf16(v16bf a, v16bf b, v8f c) {
    return __builtin_amdgcn_wmma_f32_16x16x32_bf16(false, a, false, b, (short)0, c,
                                                   false, false);
}

// ShiftedSoftplus, numerically stable
static __device__ __forceinline__ float sspf(float x) {
    float ax = __builtin_fabsf(x);
    return __builtin_fmaxf(x, 0.0f) + __logf(1.0f + __expf(-ax)) - 0.69314718055994531f;
}

// ---- WMMA fragment loaders (bf16, 16x16x32) -------------------------------

// A fragment from f32 row-major, ld = 128, padded buffer (no clamp, no guard).
static __device__ __forceinline__ v16bf load_a_f32_128(const float* __restrict__ A,
                                                       int row0, int k0) {
    int lane = threadIdx.x & 31;
    const float* rp = A + (size_t)(row0 + (lane & 15)) * HDIM + k0 + ((lane >> 4) << 3);
    float4 x0 = *(const float4*)(rp);
    float4 x1 = *(const float4*)(rp + 4);
    float4 x2 = *(const float4*)(rp + 16);
    float4 x3 = *(const float4*)(rp + 20);
    v16bf a;
    a[0]  = (__bf16)x0.x; a[1]  = (__bf16)x0.y; a[2]  = (__bf16)x0.z; a[3]  = (__bf16)x0.w;
    a[4]  = (__bf16)x1.x; a[5]  = (__bf16)x1.y; a[6]  = (__bf16)x1.z; a[7]  = (__bf16)x1.w;
    a[8]  = (__bf16)x2.x; a[9]  = (__bf16)x2.y; a[10] = (__bf16)x2.z; a[11] = (__bf16)x2.w;
    a[12] = (__bf16)x3.x; a[13] = (__bf16)x3.y; a[14] = (__bf16)x3.z; a[15] = (__bf16)x3.w;
    return a;
}

// Same, but with row clamp — for exactly-sized input arrays (xx)
static __device__ __forceinline__ v16bf load_a_f32_128c(const float* __restrict__ A,
                                                        int row0, int k0, int nrows) {
    int lane = threadIdx.x & 31;
    int r = row0 + (lane & 15);
    if (r >= nrows) r = nrows - 1;
    const float* rp = A + (size_t)r * HDIM + k0 + ((lane >> 4) << 3);
    float4 x0 = *(const float4*)(rp);
    float4 x1 = *(const float4*)(rp + 4);
    float4 x2 = *(const float4*)(rp + 16);
    float4 x3 = *(const float4*)(rp + 20);
    v16bf a;
    a[0]  = (__bf16)x0.x; a[1]  = (__bf16)x0.y; a[2]  = (__bf16)x0.z; a[3]  = (__bf16)x0.w;
    a[4]  = (__bf16)x1.x; a[5]  = (__bf16)x1.y; a[6]  = (__bf16)x1.z; a[7]  = (__bf16)x1.w;
    a[8]  = (__bf16)x2.x; a[9]  = (__bf16)x2.y; a[10] = (__bf16)x2.z; a[11] = (__bf16)x2.w;
    a[12] = (__bf16)x3.x; a[13] = (__bf16)x3.y; a[14] = (__bf16)x3.z; a[15] = (__bf16)x3.w;
    return a;
}

// Generic A fragment (scalar path, K clamp) — used for edge_attr (ld=50, no row tail)
static __device__ __forceinline__ v16bf load_a_f32(const float* __restrict__ A, int row0,
                                                   int ldA, int k0, int K) {
    int lane = threadIdx.x & 31;
    const float* rp = A + (size_t)(row0 + (lane & 15)) * ldA;
    int kh = k0 + ((lane >> 4) << 3);
    v16bf a;
#pragma unroll
    for (int i = 0; i < 8; ++i) {
        int ka = kh + i;
        int kb = kh + 16 + i;
        a[i]     = (__bf16)((ka < K) ? rp[ka] : 0.0f);
        a[i + 8] = (__bf16)((kb < K) ? rp[kb] : 0.0f);
    }
    return a;
}

// B fragment from pre-packed bf16 fragment-order weights:
// layout [col_tile][k_step][lane][16] — one aligned 32-byte load per lane.
static __device__ __forceinline__ v16bf load_b_pk(const __bf16* __restrict__ P,
                                                  int c0, int s, int S) {
    int lane = threadIdx.x & 31;
    return *(const v16bf*)(P + (size_t)((((c0 >> 4) * S + s) * 32 + lane) << 4));
}

// A from a per-wave 16x128 bf16 LDS tile (row-major) — activation staging
static __device__ __forceinline__ v16bf load_a_lds(const __bf16* __restrict__ L, int k0) {
    int lane = threadIdx.x & 31;
    const __bf16* rp = L + (lane & 15) * HDIM;
    int kh = k0 + ((lane >> 4) << 3);
    v16bf a;
#pragma unroll
    for (int i = 0; i < 8; ++i) {
        a[i]     = rp[kh + i];
        a[i + 8] = rp[kh + 16 + i];
    }
    return a;
}

// sinusoidal time embedding element (k in [0,128)) -> swish
static __device__ __forceinline__ float time_elem(float tv, int k) {
    const float lc = -0.109646909f; // -ln(1000)/63
    float e = tv * __expf((float)(k & 63) * lc);
    float v = (k >= 64) ? __cosf(e) : __sinf(e);
    return __fdividef(v, 1.0f + __expf(-v));
}

static __device__ __forceinline__ void atomic_add_f32(float* p, float v) {
    __hip_atomic_fetch_add(p, v, __ATOMIC_RELAXED, __HIP_MEMORY_SCOPE_AGENT);
}

// ---- Kernel 0: pack one weight matrix [Ksrc x 128] (f32) into bf16 --------
// fragment-order [8 col_tiles][S k_steps][32 lanes][16], zero-padding K.
__global__ __launch_bounds__(256) void k_pack(const float* __restrict__ W,
                                              __bf16* __restrict__ P, int S, int Ksrc) {
    int t = blockIdx.x * 256 + threadIdx.x;
    if (t >= 8 * S * 32 * 16) return;
    int i    = t & 15;
    int lane = (t >> 4) & 31;
    int rest = t >> 9;              // ct * S + s
    int s    = rest % S;
    int ct   = rest / S;
    int c = ct * 16 + (lane & 15);
    int k = 32 * s + ((lane >> 4) << 3) + (i & 7) + ((i >> 3) << 4);
    P[t] = (__bf16)((k < Ksrc) ? W[(size_t)k * HDIM + c] : 0.0f);
}

// ---- Kernel 1: h1 = xx @ c1_m1w ; t = swish(time_emb(tt)) @ te_w + te_b ----
// Outputs are padded buffers: stores completely unguarded.
__global__ __launch_bounds__(256) void k_node_pre(
    const float* __restrict__ xx, const float* __restrict__ tt,
    const __bf16* __restrict__ pw1, const __bf16* __restrict__ ptew,
    const float* __restrict__ teb,
    float* __restrict__ h1, float* __restrict__ tout)
{
    int wave = threadIdx.x >> 5, lane = threadIdx.x & 31;
    int hi = lane >> 4, lo = lane & 15;
    int row0 = (blockIdx.x * 8 + wave) * 16;

    v16bf a[4];
#pragma unroll
    for (int s = 0; s < 4; ++s) a[s] = load_a_f32_128c(xx, row0, 32 * s, NNODES);

    for (int c0 = 0; c0 < HDIM; c0 += 16) {
        v8f acc = {};
#pragma unroll
        for (int s = 0; s < 4; ++s)
            acc = wmma_bf16(a[s], load_b_pk(pw1, c0, s, 4), acc);
        int col = c0 + lo;
#pragma unroll
        for (int v = 0; v < 8; ++v)
            h1[(size_t)(row0 + v + 8 * hi) * HDIM + col] = acc[v];
    }

    // time embedding GEMM
    {
        int r = row0 + lo;
        if (r >= NNODES) r = NNODES - 1;
        float tv = tt[r];
#pragma unroll
        for (int s = 0; s < 4; ++s) {
            int kh = 32 * s + (hi << 3);
            v16bf at;
#pragma unroll
            for (int i = 0; i < 8; ++i) {
                at[i]     = (__bf16)time_elem(tv, kh + i);
                at[i + 8] = (__bf16)time_elem(tv, kh + 16 + i);
            }
            a[s] = at;
        }
        for (int c0 = 0; c0 < HDIM; c0 += 16) {
            v8f acc = {};
#pragma unroll
            for (int s = 0; s < 4; ++s)
                acc = wmma_bf16(a[s], load_b_pk(ptew, c0, s, 4), acc);
            int col = c0 + lo;
            float bias = teb[col];
#pragma unroll
            for (int v = 0; v < 8; ++v)
                tout[(size_t)(row0 + v + 8 * hi) * HDIM + col] = acc[v] + bias;
        }
    }
}

// ---- Kernel 2: edge MLP -> Wf (bf16, stored) ; msg1 scatter-add -----------
// NEDGES % 128 == 0: no tail anywhere in this kernel.
__global__ __launch_bounds__(256) void k_edge1(
    const float* __restrict__ eattr, const float* __restrict__ elen,
    const int* __restrict__ eidx,
    const __bf16* __restrict__ pem1, const float* __restrict__ em1b,
    const __bf16* __restrict__ pem2, const float* __restrict__ em2b,
    const float* __restrict__ h1,
    __bf16* __restrict__ wf, float* __restrict__ agg)
{
    __shared__ __bf16 lds[8 * 16 * HDIM];
    int wave = threadIdx.x >> 5, lane = threadIdx.x & 31;
    int hi = lane >> 4, lo = lane & 15;
    int e0 = (blockIdx.x * 8 + wave) * 16;
    __bf16* my = lds + wave * 16 * HDIM;

    // speculative prefetch of next block's edge_attr rows (OOB is dropped)
    __builtin_prefetch(eattr + (size_t)(e0 + 128) * ECH + lane * 16, 0, 1);

    // GEMM1: ssp(edge_attr @ em1_w + b), K=50 zero-padded to 64 at pack time
    v16bf a0 = load_a_f32(eattr, e0, ECH, 0,  ECH);
    v16bf a1 = load_a_f32(eattr, e0, ECH, 32, ECH);
    for (int c0 = 0; c0 < HDIM; c0 += 16) {
        v8f acc = {};
        acc = wmma_bf16(a0, load_b_pk(pem1, c0, 0, 2), acc);
        acc = wmma_bf16(a1, load_b_pk(pem1, c0, 1, 2), acc);
        int col = c0 + lo;
        float bias = em1b[col];
#pragma unroll
        for (int v = 0; v < 8; ++v)
            my[(v + 8 * hi) * HDIM + col] = (__bf16)sspf(acc[v] + bias);
    }
    __syncthreads();

    // GEMM2: @ em2_w + b, * cutoff ; store Wf ; gather h1[src]*Wf -> agg[dst]
    v16bf a2[4];
#pragma unroll
    for (int s = 0; s < 4; ++s) a2[s] = load_a_lds(my, 32 * s);

    const int* __restrict__ src = eidx;
    const int* __restrict__ dst = eidx + NEDGES;

    for (int c0 = 0; c0 < HDIM; c0 += 16) {
        v8f acc = {};
#pragma unroll
        for (int s = 0; s < 4; ++s)
            acc = wmma_bf16(a2[s], load_b_pk(pem2, c0, s, 4), acc);
        int col = c0 + lo;
        float bias = em2b[col];
#pragma unroll
        for (int v = 0; v < 8; ++v) {
            int e = e0 + v + 8 * hi;
            float el = elen[e];
            float C = 0.5f * (__cosf(el * 3.926990816987241f) + 1.0f); // pi/0.8
            C = (el <= 0.8f && el >= 0.0f) ? C : 0.0f;
            float w = (acc[v] + bias) * C;
            wf[(size_t)e * HDIM + col] = (__bf16)w;
            int sn = src[e], dn = dst[e];
            float m = h1[(size_t)sn * HDIM + col] * w;
            atomic_add_f32(&agg[(size_t)dn * HDIM + col], m);
        }
    }
}

// ---- Kernel 3: x = ssp(agg@c1_m2w + b + t) ; h2 = x @ c2_m1w --------------
// All buffers padded: fully branch-free.
__global__ __launch_bounds__(256) void k_node_mid(
    const float* __restrict__ agg, const float* __restrict__ tbuf,
    const __bf16* __restrict__ pw2, const float* __restrict__ b2,
    const __bf16* __restrict__ pw3, float* __restrict__ h2)
{
    __shared__ __bf16 lds[8 * 16 * HDIM];
    int wave = threadIdx.x >> 5, lane = threadIdx.x & 31;
    int hi = lane >> 4, lo = lane & 15;
    int row0 = (blockIdx.x * 8 + wave) * 16;
    __bf16* my = lds + wave * 16 * HDIM;

    v16bf a[4];
#pragma unroll
    for (int s = 0; s < 4; ++s) a[s] = load_a_f32_128(agg, row0, 32 * s);

    for (int c0 = 0; c0 < HDIM; c0 += 16) {
        v8f acc = {};
#pragma unroll
        for (int s = 0; s < 4; ++s)
            acc = wmma_bf16(a[s], load_b_pk(pw2, c0, s, 4), acc);
        int col = c0 + lo;
        float bias = b2[col];
#pragma unroll
        for (int v = 0; v < 8; ++v) {
            int r = row0 + v + 8 * hi;
            float val = sspf(acc[v] + bias + tbuf[(size_t)r * HDIM + col]);
            my[(v + 8 * hi) * HDIM + col] = (__bf16)val;
        }
    }
    __syncthreads();

#pragma unroll
    for (int s = 0; s < 4; ++s) a[s] = load_a_lds(my, 32 * s);
    for (int c0 = 0; c0 < HDIM; c0 += 16) {
        v8f acc = {};
#pragma unroll
        for (int s = 0; s < 4; ++s)
            acc = wmma_bf16(a[s], load_b_pk(pw3, c0, s, 4), acc);
        int col = c0 + lo;
#pragma unroll
        for (int v = 0; v < 8; ++v)
            h2[(size_t)(row0 + v + 8 * hi) * HDIM + col] = acc[v];
    }
}

// ---- Kernel 4: msg2 = h2[src] * Wf -> atomic agg[dst] ---------------------
// Grid is exactly NEDGES*32/256 blocks: no bounds check needed.
__global__ __launch_bounds__(256) void k_edge2(
    const __bf16* __restrict__ wf, const float* __restrict__ h2,
    const int* __restrict__ eidx, float* __restrict__ agg)
{
    int t = blockIdx.x * 256 + threadIdx.x;
    int e = t >> 5;
    int c = (t & 31) << 2;
    int sn = eidx[e], dn = eidx[NEDGES + e];
    v4bf wv = *(const v4bf*)(wf + (size_t)e * HDIM + c);
    const float4 hv = *(const float4*)(h2 + (size_t)sn * HDIM + c);
    float* ap = agg + (size_t)dn * HDIM + c;
    atomic_add_f32(ap + 0, hv.x * (float)wv[0]);
    atomic_add_f32(ap + 1, hv.y * (float)wv[1]);
    atomic_add_f32(ap + 2, hv.z * (float)wv[2]);
    atomic_add_f32(ap + 3, hv.w * (float)wv[3]);
}

// ---- Kernel 5: out = ssp( ssp(agg@c2_m2w + b) @ lin_w + lin_b ) -----------
// Only the final store targets an exactly-sized buffer; guard with a truly
// scalar (readfirstlane) wave-uniform condition.
__global__ __launch_bounds__(256) void k_node_out(
    const float* __restrict__ agg,
    const __bf16* __restrict__ pw4, const float* __restrict__ b4,
    const __bf16* __restrict__ pw5, const float* __restrict__ b5,
    float* __restrict__ out)
{
    __shared__ __bf16 lds[8 * 16 * HDIM];
    int wave = threadIdx.x >> 5, lane = threadIdx.x & 31;
    int hi = lane >> 4, lo = lane & 15;
    int row0 = (blockIdx.x * 8 + wave) * 16;
    bool full = (__builtin_amdgcn_readfirstlane(row0) + 16 <= NNODES); // scalar branch
    __bf16* my = lds + wave * 16 * HDIM;

    v16bf a[4];
#pragma unroll
    for (int s = 0; s < 4; ++s) a[s] = load_a_f32_128(agg, row0, 32 * s);

    for (int c0 = 0; c0 < HDIM; c0 += 16) {
        v8f acc = {};
#pragma unroll
        for (int s = 0; s < 4; ++s)
            acc = wmma_bf16(a[s], load_b_pk(pw4, c0, s, 4), acc);
        int col = c0 + lo;
        float bias = b4[col];
#pragma unroll
        for (int v = 0; v < 8; ++v)
            my[(v + 8 * hi) * HDIM + col] = (__bf16)sspf(acc[v] + bias);
    }
    __syncthreads();

#pragma unroll
    for (int s = 0; s < 4; ++s) a[s] = load_a_lds(my, 32 * s);
    for (int c0 = 0; c0 < HDIM; c0 += 16) {
        v8f acc = {};
#pragma unroll
        for (int s = 0; s < 4; ++s)
            acc = wmma_bf16(a[s], load_b_pk(pw5, c0, s, 4), acc);
        int col = c0 + lo;
        float bias = b5[col];
        if (full) {
#pragma unroll
            for (int v = 0; v < 8; ++v)
                out[(size_t)(row0 + v + 8 * hi) * HDIM + col] = sspf(acc[v] + bias);
        } else {
#pragma unroll
            for (int v = 0; v < 8; ++v) {
                int r = row0 + v + 8 * hi;
                if (r < NNODES) out[(size_t)r * HDIM + col] = sspf(acc[v] + bias);
            }
        }
    }
}

extern "C" void kernel_launch(void* const* d_in, const int* in_sizes, int n_in,
                              void* d_out, int out_size, void* d_ws, size_t ws_size,
                              hipStream_t stream) {
    (void)in_sizes; (void)n_in; (void)out_size; (void)ws_size;
    const float* tt    = (const float*)d_in[0];
    const float* xx    = (const float*)d_in[1];
    const int*   eidx  = (const int*)d_in[2];
    const float* elen  = (const float*)d_in[3];
    const float* eattr = (const float*)d_in[4];
    const float* em1w  = (const float*)d_in[5];
    const float* em1b  = (const float*)d_in[6];
    const float* em2w  = (const float*)d_in[7];
    const float* em2b  = (const float*)d_in[8];
    const float* c1m1w = (const float*)d_in[9];
    const float* c1m2w = (const float*)d_in[10];
    const float* c1m2b = (const float*)d_in[11];
    const float* c2m1w = (const float*)d_in[12];
    const float* c2m2w = (const float*)d_in[13];
    const float* c2m2b = (const float*)d_in[14];
    const float* tew   = (const float*)d_in[15];
    const float* teb   = (const float*)d_in[16];
    const float* linw  = (const float*)d_in[17];
    const float* linb  = (const float*)d_in[18];

    const size_t nhp = (size_t)NROWSP * HDIM;   // padded node-row buffers
    float*  h    = (float*)d_ws;        // h1, then reused as h2
    float*  tbuf = h + nhp;             // time embedding projection
    float*  agg  = tbuf + nhp;          // agg1, re-zeroed for agg2
    __bf16* wf   = (__bf16*)(agg + nhp); // E x 128 filter, bf16
    // packed bf16 weight fragments (all offsets 32B-aligned)
    const size_t PK4 = 8 * 4 * 32 * 16;   // K=128 matrices
    const size_t PK2 = 8 * 2 * 32 * 16;   // em1_w (K=50 -> 64)
    __bf16* pk   = wf + (size_t)NEDGES * HDIM;
    __bf16* pw1  = pk;              // c1_m1w
    __bf16* ptew = pw1 + PK4;       // te_w
    __bf16* pem1 = ptew + PK4;      // em1_w
    __bf16* pem2 = pem1 + PK2;      // em2_w
    __bf16* pw2  = pem2 + PK4;      // c1_m2w
    __bf16* pw3  = pw2 + PK4;       // c2_m1w
    __bf16* pw4  = pw3 + PK4;       // c2_m2w
    __bf16* pw5  = pw4 + PK4;       // lin_w

    const int nodeBlocks = NROWSP / 128;          // 391, exact with padding
    const int edgeBlocks = NEDGES / 128;
    const int pb4 = (int)((PK4 + 255) / 256), pb2 = (int)((PK2 + 255) / 256);

    // one-shot weight packing (cheap; deterministic every call)
    k_pack<<<pb4, 256, 0, stream>>>(c1m1w, pw1, 4, HDIM);
    k_pack<<<pb4, 256, 0, stream>>>(tew, ptew, 4, HDIM);
    k_pack<<<pb2, 256, 0, stream>>>(em1w, pem1, 2, ECH);
    k_pack<<<pb4, 256, 0, stream>>>(em2w, pem2, 4, HDIM);
    k_pack<<<pb4, 256, 0, stream>>>(c1m2w, pw2, 4, HDIM);
    k_pack<<<pb4, 256, 0, stream>>>(c2m1w, pw3, 4, HDIM);
    k_pack<<<pb4, 256, 0, stream>>>(c2m2w, pw4, 4, HDIM);
    k_pack<<<pb4, 256, 0, stream>>>(linw, pw5, 4, HDIM);

    hipMemsetAsync(agg, 0, nhp * sizeof(float), stream);
    k_node_pre<<<nodeBlocks, 256, 0, stream>>>(xx, tt, pw1, ptew, teb, h, tbuf);
    k_edge1<<<edgeBlocks, 256, 0, stream>>>(eattr, elen, eidx, pem1, em1b, pem2, em2b,
                                            h, wf, agg);
    k_node_mid<<<nodeBlocks, 256, 0, stream>>>(agg, tbuf, pw2, c1m2b, pw3, h);
    hipMemsetAsync(agg, 0, nhp * sizeof(float), stream);
    k_edge2<<<(NEDGES * 32) / 256, 256, 0, stream>>>(wf, h, eidx, agg);
    k_node_out<<<nodeBlocks, 256, 0, stream>>>(agg, pw4, c2m2b, pw5, linb,
                                               (float*)d_out);
}
